// LinearAttention_14491219656765
// MI455X (gfx1250) — matrix operand order repaired and checked by reading, not compiled, not run on hardware
//
#include <hip/hip_runtime.h>

// ---------------------------------------------------------------------------
// LinearAttention fused pipeline for gfx1250 (CDNA5).
// bf16 WMMA 16x16x32 + async global->LDS copies (ASYNCcnt) for x staging.
// b=8, c=64, n=32768, HEADS=4, DIM_HEAD=32.
// ---------------------------------------------------------------------------

typedef __attribute__((ext_vector_type(16))) __bf16 bf16x16;
typedef __attribute__((ext_vector_type(8)))  __bf16 bf16x8;
typedef __attribute__((ext_vector_type(8)))  float  f32x8;

#define B_      8
#define C_      64
#define NTOK    32768
#define HEADS_  4
#define DH      32
#define CHUNK   64      // tokens per inner chunk
#define NCHUNK  4       // chunks per workgroup (256 tokens)
#define SCALE_  0.17677669529663687f   // 32^-0.5
#define EPS_    1e-5f

// ---------------------------------------------------------------------------
// fragment helpers
// ---------------------------------------------------------------------------
__device__ __forceinline__ bf16x16 frag_join(const __bf16* p0, const __bf16* p1) {
    bf16x8 lo = *(const bf16x8*)p0;
    bf16x8 hi = *(const bf16x8*)p1;
    return __builtin_shufflevector(lo, hi, 0, 1, 2, 3, 4, 5, 6, 7,
                                           8, 9, 10, 11, 12, 13, 14, 15);
}

// A fragment (16x32, MxK), row-major source, row stride in bf16 elems.
// ISA: lanes 0-15 row M=lane, K {0..7,16..23}; lanes 16-31 K {8..15,24..31}.
__device__ __forceinline__ bf16x16 load_a(const __bf16* src, int stride,
                                          int m0, int k0, int lane) {
    const __bf16* r = src + (size_t)(m0 + (lane & 15)) * stride
                          + k0 + ((lane >> 4) << 3);
    return frag_join(r, r + 16);
}

// B fragment (32x16, KxN), column-major storage srcT[n][k].
// ISA: lane holds column N=lane&15; lanes 0-15 K {0..7,16..23}; lanes 16-31 K {8..15,24..31}.
__device__ __forceinline__ bf16x16 load_b(const __bf16* srcT, int stride,
                                          int n0, int k0, int lane) {
    const __bf16* r = srcT + (size_t)(n0 + (lane & 15)) * stride
                           + k0 + ((lane >> 4) << 3);
    return frag_join(r, r + 16);
}

__device__ __forceinline__ f32x8 wmma_bf16(bf16x16 a, bf16x16 b, f32x8 c) {
    return __builtin_amdgcn_wmma_f32_16x16x32_bf16(false, a, false, b,
                                                   (short)0, c, false, false);
}

// ---------------------------------------------------------------------------
// async global->LDS copy (CDNA5, ASYNCcnt-tracked)
// LDS offset = low 32 bits of the generic pointer (aperture: addr[31:0]).
// ---------------------------------------------------------------------------
__device__ __forceinline__ unsigned lds_off(const void* p) {
    return (unsigned)(uintptr_t)p;
}
__device__ __forceinline__ void async_copy_b128(unsigned dst_lds,
                                                const void* src_global) {
    asm volatile("global_load_async_to_lds_b128 %0, %1, off"
                 :: "v"(dst_lds), "v"(src_global) : "memory");
}
__device__ __forceinline__ void wait_async0() {
    asm volatile("s_wait_asynccnt 0" ::: "memory");
}

// stage one 64-token x-chunk: async copy f32 [c][tok] into stg, then
// transpose-convert to bf16 xT[tok][c]
__device__ __forceinline__ void stage_x_chunk(const float* __restrict__ xb,
                                              int t0, int tid,
                                              float (*stg)[CHUNK],
                                              __bf16 (*xT)[72]) {
    #pragma unroll
    for (int j = 0; j < 4; ++j) {
        int idx = tid + j * 256;            // 1024 b128 transfers total
        int c = idx >> 4, seg = idx & 15;   // 16 x b128 per channel row
        async_copy_b128(lds_off(&stg[c][seg * 4]),
                        xb + (size_t)c * NTOK + t0 + seg * 4);
    }
    wait_async0();
    __syncthreads();
    #pragma unroll
    for (int i = 0; i < 16; ++i) {
        int idx = tid + i * 256;
        int c = idx >> 6, tok = idx & 63;
        xT[tok][c] = (__bf16)stg[c][tok];
    }
    __syncthreads();
}

// ---------------------------------------------------------------------------
// K0: zero accumulators, convert weights to bf16 into workspace
// ---------------------------------------------------------------------------
__global__ void la_k0_init(const float* __restrict__ w_qkv,
                           const float* __restrict__ w_out,
                           __bf16* __restrict__ w_bf, __bf16* __restrict__ wout_bf,
                           float* __restrict__ ctx_raw, float* __restrict__ rowsum,
                           float* __restrict__ stats) {
    int i = blockIdx.x * 256 + threadIdx.x;
    int stride = gridDim.x * 256;
    for (int j = i; j < 384 * 64; j += stride) w_bf[j] = (__bf16)w_qkv[j];
    for (int j = i; j < 64 * 128; j += stride) wout_bf[j] = (__bf16)w_out[j];
    for (int j = i; j < B_ * HEADS_ * DH * DH; j += stride) ctx_raw[j] = 0.f;
    for (int j = i; j < B_ * HEADS_ * DH; j += stride) rowsum[j] = 0.f;
    for (int j = i; j < 2 * B_; j += stride) stats[j] = 0.f;
}

// ---------------------------------------------------------------------------
// K1: streaming context accumulation.
//   k,v = w_qkv[128:384] @ x_tile (WMMA) -> LDS (exp applied to k),
//   ctx_raw += exp(k) @ v^T (WMMA), rowsum += sum_n exp(k). Atomic flush.
// ---------------------------------------------------------------------------
__global__ __launch_bounds__(256) void la_k1_context(
        const float* __restrict__ x, const __bf16* __restrict__ w_bf,
        float* __restrict__ ctx_raw, float* __restrict__ rowsum) {
    __shared__ __align__(16) float  stg[64][CHUNK];   // f32 staging [c][tok] 16 KB
    __shared__ __align__(16) __bf16 xT [CHUNK][72];   // [tok][c]   (B operand)
    __shared__ __align__(16) __bf16 ekT[128][72];     // exp(k):[h*32+d][tok] (A)
    __shared__ __align__(16) __bf16 vT [128][72];     // v:[h*32+e][tok] (B)

    const int tid  = threadIdx.x;
    const int lane = tid & 31;
    const int wave = tid >> 5;
    const int b    = blockIdx.y;
    const int nb   = blockIdx.x * (CHUNK * NCHUNK);
    const float* xb = x + (size_t)b * C_ * NTOK;

    // hoisted A fragments of w_kv (chunk-invariant; wave covers mt = 2w, 2w+1)
    bf16x16 a_kv[2][2];
    #pragma unroll
    for (int m = 0; m < 2; ++m)
        #pragma unroll
        for (int ks = 0; ks < 2; ++ks)
            a_kv[m][ks] = load_a(w_bf + 128 * 64, 64,
                                 (wave * 2 + m) * 16, ks * 32, lane);

    f32x8 ctx_acc[2] = {};
    float rs_acc = 0.f;

    const int h_s2  = wave >> 1;   // stage-2 head (wave-constant)
    const int mt_s2 = wave & 1;    // stage-2 d-tile (wave-constant)

    for (int ch = 0; ch < NCHUNK; ++ch) {
        stage_x_chunk(xb, nb + ch * CHUNK, tid, stg, xT);

        // stage 1: kv = w_kv @ x_tile ; 16 Mtiles x 4 Ntiles, 8 tiles/wave
        #pragma unroll
        for (int i = 0; i < 8; ++i) {
            int msel = i >> 2;                 // 0..1 -> mt = wave*2+msel
            int nt   = i & 3;
            f32x8 acc = {};
            #pragma unroll
            for (int ks = 0; ks < 2; ++ks) {
                bf16x16 bb = load_b(&xT[0][0], 72, nt * 16, ks * 32, lane);
                acc = wmma_bf16(a_kv[msel][ks], bb, acc);
            }
            int row0 = (wave * 2 + msel) * 16 + ((lane >> 4) << 3);
            int col  = nt * 16 + (lane & 15);
            if (wave < 4) {        // k rows (wave-uniform branch) -> exp(k)
                #pragma unroll
                for (int r = 0; r < 8; ++r)
                    ekT[row0 + r][col] = (__bf16)__expf(acc[r]);
            } else {               // v rows
                #pragma unroll
                for (int r = 0; r < 8; ++r)
                    vT[row0 - 128 + r][col] = (__bf16)acc[r];
            }
        }
        __syncthreads();

        // rowsum partial (pure reads, overlaps stage-2 reads)
        if (tid < 128) {
            float s = 0.f;
            for (int j = 0; j < CHUNK; ++j) s += (float)ekT[tid][j];
            rs_acc += s;
        }

        // stage 2: ctx += exp_k @ v^T ; 16 tiles, 2/wave (nt = i)
        #pragma unroll
        for (int i = 0; i < 2; ++i) {
            #pragma unroll
            for (int ks = 0; ks < 2; ++ks) {
                bf16x16 a  = load_a(&ekT[h_s2 * 32][0], 72, mt_s2 * 16,
                                    ks * 32, lane);
                bf16x16 bb = load_b(&vT[h_s2 * 32][0], 72, i * 16,
                                    ks * 32, lane);
                ctx_acc[i] = wmma_bf16(a, bb, ctx_acc[i]);
            }
        }
        __syncthreads();
    }

    // flush context tiles + rowsum with f32 atomics
    #pragma unroll
    for (int i = 0; i < 2; ++i) {
        int d0 = mt_s2 * 16 + ((lane >> 4) << 3);
        int e  = i * 16 + (lane & 15);
        float* dst = ctx_raw + ((size_t)(b * HEADS_ + h_s2)) * DH * DH;
        #pragma unroll
        for (int r = 0; r < 8; ++r)
            atomicAdd(&dst[(d0 + r) * DH + e], ctx_acc[i][r]);
    }
    if (tid < 128) atomicAdd(&rowsum[b * 128 + tid], rs_acc);
}

// ---------------------------------------------------------------------------
// K2: ctx = SCALE * ctx_raw / rowsum, stored transposed ([b][h][e][d]) as bf16
// ---------------------------------------------------------------------------
__global__ void la_k2_ctx(const float* __restrict__ ctx_raw,
                          const float* __restrict__ rowsum,
                          __bf16* __restrict__ ctxT) {
    int idx = blockIdx.x * 256 + threadIdx.x;
    if (idx >= B_ * HEADS_ * DH * DH) return;
    int e = idx & 31, d = (idx >> 5) & 31, bh = idx >> 10;
    float v = SCALE_ * ctx_raw[idx] / rowsum[bh * DH + d];
    ctxT[(size_t)bh * DH * DH + e * DH + d] = (__bf16)v;
}

// ---------------------------------------------------------------------------
// K3: q = w_q @ x (WMMA) -> softmax over D -> out = ctx^T @ q_soft (WMMA)
//     -> y = w_out @ out + b_out (WMMA) -> d_out, accumulate GN stats.
// ---------------------------------------------------------------------------
__global__ __launch_bounds__(256) void la_k3_out(
        const float* __restrict__ x, const __bf16* __restrict__ w_bf,
        const __bf16* __restrict__ wout_bf, const __bf16* __restrict__ ctxT,
        const float* __restrict__ b_out, float* __restrict__ y,
        float* __restrict__ stats) {
    __shared__ __align__(16) float  stg[64][CHUNK];   // f32 staging 16 KB
    __shared__ __align__(16) __bf16 xT[CHUNK][72];    // [tok][c]
    __shared__ __align__(16) __bf16 qT[CHUNK][136];   // [tok][qrow 0..127]
    __shared__ __align__(16) __bf16 oT[CHUNK][136];   // [tok][outrow 0..127]

    const int tid  = threadIdx.x;
    const int lane = tid & 31;
    const int wave = tid >> 5;
    const int b    = blockIdx.y;
    const int nb   = blockIdx.x * (CHUNK * NCHUNK);
    const float* xb = x + (size_t)b * C_ * NTOK;

    // hoisted chunk-invariant A fragments
    bf16x16 a_q[2];                        // q-gemm: mt = wave
    #pragma unroll
    for (int ks = 0; ks < 2; ++ks)
        a_q[ks] = load_a(w_bf, 64, wave * 16, ks * 32, lane);

    const int h3  = wave >> 1;             // ctx-gemm head (wave-constant)
    const int mt3 = wave & 1;              // ctx-gemm e-tile (wave-constant)
    bf16x16 a_ctx = load_a(ctxT + (size_t)(b * HEADS_ + h3) * DH * DH,
                           32, mt3 * 16, 0, lane);

    const int mt4 = wave >> 1;             // final-gemm row tile (wave-constant)
    bf16x16 a_o[4];
    #pragma unroll
    for (int ks = 0; ks < 4; ++ks)
        a_o[ks] = load_a(wout_bf, 128, mt4 * 16, ks * 32, lane);

    float s1 = 0.f, s2 = 0.f;

    for (int ch = 0; ch < NCHUNK; ++ch) {
        const int t0 = nb + ch * CHUNK;
        stage_x_chunk(xb, t0, tid, stg, xT);

        // q = w_q @ x : 8 Mtiles x 4 Ntiles, 4 tiles/wave (nt = i); token-major
        #pragma unroll
        for (int i = 0; i < 4; ++i) {
            f32x8 acc = {};
            #pragma unroll
            for (int ks = 0; ks < 2; ++ks) {
                bf16x16 bb = load_b(&xT[0][0], 72, i * 16, ks * 32, lane);
                acc = wmma_bf16(a_q[ks], bb, acc);
            }
            int tok  = i * 16 + (lane & 15);
            int row0 = wave * 16 + ((lane >> 4) << 3);
            #pragma unroll
            for (int r = 0; r < 8; ++r)
                qT[tok][row0 + r] = (__bf16)acc[r];
        }
        __syncthreads();

        // softmax over D within each head: 4 heads x 64 tokens = 256 threads
        {
            int h = tid >> 6, tok = tid & 63;
            __bf16* p = &qT[tok][h * 32];
            float v[32], m = -1e30f;
            #pragma unroll
            for (int j = 0; j < 32; ++j) { v[j] = (float)p[j]; m = fmaxf(m, v[j]); }
            float s = 0.f;
            #pragma unroll
            for (int j = 0; j < 32; ++j) { v[j] = __expf(v[j] - m); s += v[j]; }
            float inv = 1.f / s;
            #pragma unroll
            for (int j = 0; j < 32; ++j) p[j] = (__bf16)(v[j] * inv);
        }
        __syncthreads();

        // out = ctx^T @ q_soft : 4h x 2mt x 4nt tiles, 4/wave (nt = i)
        #pragma unroll
        for (int i = 0; i < 4; ++i) {
            bf16x16 bb = load_b(&qT[0][0] + h3 * 32, 136, i * 16, 0, lane);
            f32x8 acc = {};
            acc = wmma_bf16(a_ctx, bb, acc);
            int tok  = i * 16 + (lane & 15);
            int row0 = h3 * 32 + mt3 * 16 + ((lane >> 4) << 3);
            #pragma unroll
            for (int r = 0; r < 8; ++r)
                oT[tok][row0 + r] = (__bf16)acc[r];
        }
        __syncthreads();

        // y = w_out @ out + b_out : 4x4 tiles, 2/wave, K=128
        #pragma unroll
        for (int i = 0; i < 2; ++i) {
            int nt = 2 * (wave & 1) + i;
            f32x8 acc = {};
            #pragma unroll
            for (int ks = 0; ks < 4; ++ks) {
                bf16x16 bb = load_b(&oT[0][0], 136, nt * 16, ks * 32, lane);
                acc = wmma_bf16(a_o[ks], bb, acc);
            }
            int tokg = t0 + nt * 16 + (lane & 15);
            int row0 = mt4 * 16 + ((lane >> 4) << 3);
            #pragma unroll
            for (int r = 0; r < 8; ++r) {
                int row = row0 + r;
                float val = acc[r] + b_out[row];
                y[(size_t)(b * C_ + row) * NTOK + tokg] = val;
                s1 += val;
                s2 += val * val;
            }
        }
        __syncthreads();
    }

    atomicAdd(&stats[b * 2 + 0], s1);
    atomicAdd(&stats[b * 2 + 1], s2);
}

// ---------------------------------------------------------------------------
// K4: in-place GroupNorm(1, 64) over d_out, per batch, + gamma/beta
// ---------------------------------------------------------------------------
__global__ void la_k4_gn(float* __restrict__ y, const float* __restrict__ stats,
                         const float* __restrict__ gamma,
                         const float* __restrict__ beta) {
    const float cnt = (float)C_ * (float)NTOK;
    const size_t total = (size_t)B_ * C_ * NTOK;
    const size_t gstride = (size_t)gridDim.x * 256;
    for (size_t i = (size_t)blockIdx.x * 256 + threadIdx.x; i < total;
         i += gstride) {
        __builtin_prefetch(&y[i + gstride], 0, 1);
        int b = (int)(i / ((size_t)C_ * NTOK));
        int c = (int)((i / NTOK) & (C_ - 1));
        float mu  = stats[b * 2 + 0] / cnt;
        float var = stats[b * 2 + 1] / cnt - mu * mu;
        float v = (y[i] - mu) * rsqrtf(var + EPS_);
        y[i] = v * gamma[c] + beta[c];
    }
}

// ---------------------------------------------------------------------------
// launcher
// ---------------------------------------------------------------------------
extern "C" void kernel_launch(void* const* d_in, const int* in_sizes, int n_in,
                              void* d_out, int out_size, void* d_ws, size_t ws_size,
                              hipStream_t stream) {
    (void)in_sizes; (void)n_in; (void)out_size; (void)ws_size;
    const float* x     = (const float*)d_in[0];
    const float* w_qkv = (const float*)d_in[1];
    const float* w_out = (const float*)d_in[2];
    const float* b_out = (const float*)d_in[3];
    const float* gamma = (const float*)d_in[4];
    const float* beta  = (const float*)d_in[5];
    float* y = (float*)d_out;

    char* ws = (char*)d_ws;
    __bf16* w_bf    = (__bf16*)(ws);            // 384*64 bf16   = 49152 B
    __bf16* wout_bf = (__bf16*)(ws + 49152);    // 64*128 bf16   = 16384 B
    __bf16* ctxT    = (__bf16*)(ws + 65536);    // 8*4*32*32 bf16= 65536 B
    float*  ctx_raw = (float*) (ws + 81920);    // 32768 f32     = 131072 B
    float*  rowsum  = (float*) (ws + 212992);   // 1024 f32      = 4096 B
    float*  stats   = (float*) (ws + 217088);   // 16 f32

    la_k0_init<<<64, 256, 0, stream>>>(w_qkv, w_out, w_bf, wout_bf,
                                       ctx_raw, rowsum, stats);
    la_k1_context<<<dim3(NTOK / (CHUNK * NCHUNK), B_), 256, 0, stream>>>(
        x, w_bf, ctx_raw, rowsum);
    la_k2_ctx<<<(B_ * HEADS_ * DH * DH + 255) / 256, 256, 0, stream>>>(
        ctx_raw, rowsum, ctxT);
    la_k3_out<<<dim3(NTOK / (CHUNK * NCHUNK), B_), 256, 0, stream>>>(
        x, w_bf, wout_bf, ctxT, b_out, y, stats);
    la_k4_gn<<<4096, 256, 0, stream>>>(y, stats, gamma, beta);
}
